// FlashAttention_31688268710280
// MI455X (gfx1250) — compile-verified, matrix operand order checked by reading, not run on hardware
//
#include <hip/hip_runtime.h>

// Flash attention forward for B=4, S=2048, H=16, D=64, causal, fp32 I/O,
// additive bias [1,H,S,S]. f16 WMMA matmuls with fp32 accumulate + fp32 softmax.

#define B_ 4
#define S_ 2048
#define H_ 16
#define D_ 64

constexpr int BLOCK_M = 64;     // q rows per workgroup (16 per wave)
constexpr int BLOCK_N = 64;     // kv rows per tile
constexpr int LDK     = 72;     // halves per LDS row (64 + 8 pad -> conflict-free b128)
constexpr float SCALE   = 0.125f;   // 1/sqrt(64), folded into Q->f16 conversion
constexpr float NEG_INF = -1e30f;

typedef __attribute__((ext_vector_type(16))) _Float16 v16h;
typedef __attribute__((ext_vector_type(8)))  _Float16 v8h;
typedef __attribute__((ext_vector_type(4)))  _Float16 v4h;
typedef __attribute__((ext_vector_type(8)))  float    v8f;
typedef __attribute__((ext_vector_type(4)))  float    v4f;

// A/B fragment layout for V_WMMA_F32_16X16X32_F16 (per ISA 7.12.2):
// lane (0..15) holds row/col (lane&15) with K-chunks {8*hi+0..7, 16+8*hi+0..7}
// of each 32-wide K block; chunks are 8 contiguous elements (16B).
__device__ inline v16h frag_from_f32(const float* __restrict__ p, float scale) {
  v4f x0 = *(const v4f*)(p +  0);
  v4f x1 = *(const v4f*)(p +  4);
  v4f x2 = *(const v4f*)(p + 16);
  v4f x3 = *(const v4f*)(p + 20);
  v16h f;
#pragma unroll
  for (int j = 0; j < 4; ++j) {
    f[j]      = (_Float16)(x0[j] * scale);
    f[4 + j]  = (_Float16)(x1[j] * scale);
    f[8 + j]  = (_Float16)(x2[j] * scale);
    f[12 + j] = (_Float16)(x3[j] * scale);
  }
  return f;
}

__device__ inline v16h frag_from_lds(const _Float16* p) {
  v8h a = *(const v8h*)(p);        // 16B aligned (row stride 144B, offsets 16B)
  v8h b = *(const v8h*)(p + 16);
  v16h f;
#pragma unroll
  for (int j = 0; j < 8; ++j) { f[j] = a[j]; f[8 + j] = b[j]; }
  return f;
}

__global__ __launch_bounds__(128)
void fa_fwd_wmma(const float* __restrict__ q, const float* __restrict__ k,
                 const float* __restrict__ v, const float* __restrict__ bias,
                 float* __restrict__ out) {
  __shared__ _Float16 Ksh[BLOCK_N * LDK];   // [kv][d]   row-major f16
  __shared__ _Float16 Vsh[D_ * LDK];        // [d][kv]   transposed f16
  __shared__ _Float16 Psh[4 * 16 * LDK];    // per-wave P tiles [m][kv]

  const int qt   = blockIdx.x;              // q tile index
  const int h    = blockIdx.y;
  const int b    = blockIdx.z;
  const int tid  = threadIdx.x;
  const int wv   = tid >> 5;                // wave 0..3
  const int lane = tid & 31;
  const int nlo  = lane & 15;
  const int hi   = lane >> 4;

  const int q0 = qt * BLOCK_M;

  // ---- Q fragments for this wave's 16 rows (scale folded in) ----
  const int qrow = q0 + 16 * wv + nlo;
  const float* qp = q + (((size_t)b * S_ + qrow) * H_ + h) * D_;
  const v16h qa0 = frag_from_f32(qp + 0  + 8 * hi, SCALE);
  const v16h qa1 = frag_from_f32(qp + 32 + 8 * hi, SCALE);

  // ---- running state: rows (8*hi + r) for r = 0..7 ----
  v8f o[4];
  float m_i[8], l_i[8];
#pragma unroll
  for (int u = 0; u < 4; ++u) o[u] = (v8f){};
#pragma unroll
  for (int r = 0; r < 8; ++r) { m_i[r] = -3.0e38f; l_i[r] = 0.0f; }

  _Float16* pw = &Psh[wv * 16 * LDK];
  const float* bb_base =
      bias + ((size_t)h * S_ + (q0 + 16 * wv + 8 * hi)) * S_;

  const int n_tiles = qt + 1;               // causal: kv tiles 0..qt
  for (int it = 0; it < n_tiles; ++it) {
    const int kv0 = it * BLOCK_N;

    // ---- cooperative stage of K (row-major) and V (transposed) into LDS ----
    __syncthreads();
#pragma unroll
    for (int i = 0; i < 8; ++i) {
      const int idx = i * 128 + tid;        // 1024 float4s of the 64x64 tile
      const int row = idx >> 4;             // kv row in tile
      const int c4  = idx & 15;             // float4 column
      const size_t g = (((size_t)b * S_ + kv0 + row) * H_ + h) * D_ + c4 * 4;
      v4f kx = *(const v4f*)(k + g);
      v4h kh;
#pragma unroll
      for (int j = 0; j < 4; ++j) kh[j] = (_Float16)kx[j];
      *(v4h*)(&Ksh[row * LDK + c4 * 4]) = kh;       // ds_store_b64
      v4f vx = *(const v4f*)(v + g);
#pragma unroll
      for (int j = 0; j < 4; ++j)
        Vsh[(c4 * 4 + j) * LDK + row] = (_Float16)vx[j];  // transpose
    }
    __syncthreads();

    const float* bb = bb_base + kv0;
    if (it + 1 < n_tiles) {                 // prefetch next tile's bias stream
#pragma unroll
      for (int r = 0; r < 8; ++r)
        __builtin_prefetch(bb + (size_t)r * S_ + BLOCK_N, 0, 1);
    }

    // ---- S = Q*K^T (+bias as initial C) : 4 N-tiles x 2 K-steps ----
    v8f st[4];
#pragma unroll
    for (int t = 0; t < 4; ++t) {
      v8f c;
#pragma unroll
      for (int r = 0; r < 8; ++r) c[r] = bb[(size_t)r * S_ + 16 * t + nlo];
      const _Float16* krow = &Ksh[(16 * t + nlo) * LDK + 8 * hi];
      v16h kb0 = frag_from_lds(krow + 0);
      v16h kb1 = frag_from_lds(krow + 32);
      c = __builtin_amdgcn_wmma_f32_16x16x32_f16(false, qa0, false, kb0,
                                                 (short)0, c, false, false);
      c = __builtin_amdgcn_wmma_f32_16x16x32_f16(false, qa1, false, kb1,
                                                 (short)0, c, false, false);
      st[t] = c;
    }

    // ---- causal mask on the diagonal tile ----
    if (kv0 == q0) {
#pragma unroll
      for (int t = 0; t < 4; ++t)
#pragma unroll
        for (int r = 0; r < 8; ++r)
          if (16 * t + nlo > 16 * wv + 8 * hi + r) st[t][r] = NEG_INF;
    }

    // ---- online softmax: row max across 16 lanes of this half-wave ----
    float mx[8];
#pragma unroll
    for (int r = 0; r < 8; ++r)
      mx[r] = fmaxf(fmaxf(st[0][r], st[1][r]), fmaxf(st[2][r], st[3][r]));
#pragma unroll
    for (int off = 8; off >= 1; off >>= 1)
#pragma unroll
      for (int r = 0; r < 8; ++r)
        mx[r] = fmaxf(mx[r], __shfl_xor(mx[r], off, 32));

    float al[8], rs[8];
#pragma unroll
    for (int r = 0; r < 8; ++r) {
      const float mn = fmaxf(m_i[r], mx[r]);
      al[r] = __expf(m_i[r] - mn);
      m_i[r] = mn;
      rs[r] = 0.0f;
    }

    // ---- P = exp(S - m); spill to per-wave LDS to re-layout C -> A frag ----
#pragma unroll
    for (int t = 0; t < 4; ++t)
#pragma unroll
      for (int r = 0; r < 8; ++r) {
        const float p = __expf(st[t][r] - m_i[r]);
        rs[r] += p;
        pw[(8 * hi + r) * LDK + 16 * t + nlo] = (_Float16)p;
      }
#pragma unroll
    for (int off = 8; off >= 1; off >>= 1)
#pragma unroll
      for (int r = 0; r < 8; ++r) rs[r] += __shfl_xor(rs[r], off, 32);
#pragma unroll
    for (int r = 0; r < 8; ++r) l_i[r] = l_i[r] * al[r] + rs[r];

    // ---- O = O*alpha + P*V ---- (same-wave DS ops are in-order)
    const v16h pa0 = frag_from_lds(&pw[nlo * LDK + 0  + 8 * hi]);
    const v16h pa1 = frag_from_lds(&pw[nlo * LDK + 32 + 8 * hi]);
#pragma unroll
    for (int u = 0; u < 4; ++u) {
      v8f oc = o[u];
#pragma unroll
      for (int r = 0; r < 8; ++r) oc[r] *= al[r];
      const _Float16* vrow = &Vsh[(16 * u + nlo) * LDK + 8 * hi];
      v16h vb0 = frag_from_lds(vrow + 0);
      v16h vb1 = frag_from_lds(vrow + 32);
      oc = __builtin_amdgcn_wmma_f32_16x16x32_f16(false, pa0, false, vb0,
                                                  (short)0, oc, false, false);
      oc = __builtin_amdgcn_wmma_f32_16x16x32_f16(false, pa1, false, vb1,
                                                  (short)0, oc, false, false);
      o[u] = oc;
    }
  }

  // ---- finalize: O / l, store fp32 [B,S,H,D] ----
  float inv_l[8];
#pragma unroll
  for (int r = 0; r < 8; ++r) inv_l[r] = 1.0f / l_i[r];  // causal => l > 0
  float* op = out + (((size_t)b * S_ + (q0 + 16 * wv + 8 * hi)) * H_ + h) * D_;
#pragma unroll
  for (int u = 0; u < 4; ++u)
#pragma unroll
    for (int r = 0; r < 8; ++r)
      op[(size_t)r * H_ * D_ + 16 * u + nlo] = o[u][r] * inv_l[r];
}

extern "C" void kernel_launch(void* const* d_in, const int* in_sizes, int n_in,
                              void* d_out, int out_size, void* d_ws, size_t ws_size,
                              hipStream_t stream) {
  (void)in_sizes; (void)n_in; (void)out_size; (void)d_ws; (void)ws_size;
  const float* q    = (const float*)d_in[0];
  const float* k    = (const float*)d_in[1];
  const float* v    = (const float*)d_in[2];
  const float* bias = (const float*)d_in[3];
  float* out        = (float*)d_out;
  dim3 grid(S_ / BLOCK_M, H_, B_);
  fa_fwd_wmma<<<grid, 128, 0, stream>>>(q, k, v, bias, out);
}
